// MoE_51144470561266
// MI455X (gfx1250) — matrix operand order repaired and checked by reading, not compiled, not run on hardware
//
#include <hip/hip_runtime.h>
#include <hip/hip_bf16.h>
#include <math.h>

// ---------------- problem constants ----------------
#define D_H   1024          // hidden
#define D_E   512           // expert width
#define N_EXP 16
#define T_TOK 8192          // 4 * 2048 tokens
#define D_SE  1024          // shared expert width (512*2)

typedef __attribute__((ext_vector_type(16))) __bf16 v16bf;
typedef __attribute__((ext_vector_type(8)))  float  v8f;

struct Frag { union { v16bf v; uint4 q[2]; }; };

__device__ __forceinline__ unsigned short f32_to_bf16_bits(float f) {
  unsigned u = __builtin_bit_cast(unsigned, f);
  u += 0x7FFFu + ((u >> 16) & 1u);      // round-to-nearest-even
  return (unsigned short)(u >> 16);
}
__device__ __forceinline__ __bf16 f32_to_bf16(float f) {
  unsigned short h = f32_to_bf16_bits(f);
  return __builtin_bit_cast(__bf16, h);
}

__device__ __forceinline__ v8f wmma_bf16(v16bf a, v16bf b, v8f c) {
  return __builtin_amdgcn_wmma_f32_16x16x32_bf16(
      false, a, false, b, (short)0, c, false, false);
}

// 16B-chunk XOR swizzle inside a row: avoids the 2KB-row-stride bank collision
__device__ __forceinline__ int sw_idx(int row, int col, int strideElems) {
  int chunk = col >> 3;          // 8 bf16 = 16B
  int within = col & 7;
  int sch = chunk ^ (row & 7);
  return row * strideElems + (sch << 3) + within;
}

// A fragment (LDS, swizzled): row m, K base k (mult of 32), half = lane>>4
__device__ __forceinline__ v16bf load_a_sw(const __bf16* s, int row, int stride,
                                           int k, int half) {
  Frag f;
  f.q[0] = *(const uint4*)(s + sw_idx(row, k + 16 * half, stride));
  f.q[1] = *(const uint4*)(s + sw_idx(row, k + 16 * half + 8, stride));
  return f.v;
}
// B fragment: p -> W[n]*Kdim + k + 16*half  (32 contiguous bf16 bytes)
__device__ __forceinline__ v16bf load_b_frag(const __bf16* p) {
  Frag f;
  f.q[0] = *(const uint4*)(p);
  f.q[1] = *(const uint4*)(p + 8);
  return f.v;
}

__device__ __forceinline__ float silu(float g) {
  return g / (1.0f + __expf(-g));
}

// ---------------- fp32 -> bf16 weight conversion ----------------
__global__ __launch_bounds__(256) void cvt_bf16_kernel(const float* __restrict__ src,
                                                       __bf16* __restrict__ dst, int n4) {
  int i = blockIdx.x * 256 + threadIdx.x;
  if (i >= n4) return;
  float4 v = ((const float4*)src)[i];
  ushort4 o;
  o.x = f32_to_bf16_bits(v.x);
  o.y = f32_to_bf16_bits(v.y);
  o.z = f32_to_bf16_bits(v.z);
  o.w = f32_to_bf16_bits(v.w);
  ((ushort4*)dst)[i] = o;
}

// ---------------- router: scores -> softmax -> top2 ----------------
__global__ __launch_bounds__(256) void router_kernel(const float* __restrict__ x,
                                                     const float* __restrict__ wr,
                                                     int* __restrict__ topk_id,
                                                     float* __restrict__ topk_w) {
  const int lane = threadIdx.x & 31;
  const int wid  = threadIdx.x >> 5;
  const int t    = blockIdx.x * 8 + wid;
  const float* xr = x + (size_t)t * D_H;

  float acc[N_EXP];
#pragma unroll
  for (int e = 0; e < N_EXP; ++e) acc[e] = 0.0f;
  for (int h = lane; h < D_H; h += 32) {
    float xv = xr[h];
#pragma unroll
    for (int e = 0; e < N_EXP; ++e) acc[e] = fmaf(xv, wr[e * D_H + h], acc[e]);
  }
#pragma unroll
  for (int e = 0; e < N_EXP; ++e) {
#pragma unroll
    for (int off = 16; off > 0; off >>= 1) acc[e] += __shfl_xor(acc[e], off, 32);
  }
  float mx = acc[0];
#pragma unroll
  for (int e = 1; e < N_EXP; ++e) mx = fmaxf(mx, acc[e]);
  float denom = 0.0f;
#pragma unroll
  for (int e = 0; e < N_EXP; ++e) denom += __expf(acc[e] - mx);
  int i0 = 0; float v0 = acc[0];
#pragma unroll
  for (int e = 1; e < N_EXP; ++e) if (acc[e] > v0) { v0 = acc[e]; i0 = e; }
  int i1 = -1; float v1 = -3.4e38f;
#pragma unroll
  for (int e = 0; e < N_EXP; ++e)
    if (e != i0 && acc[e] > v1) { v1 = acc[e]; i1 = e; }
  if (lane == 0) {
    float inv = 1.0f / denom;
    topk_id[t * 2 + 0] = i0;
    topk_id[t * 2 + 1] = i1;
    topk_w[t * 2 + 0] = __expf(v0 - mx) * inv;
    topk_w[t * 2 + 1] = __expf(v1 - mx) * inv;
  }
}

// ---------------- deterministic per-expert token list build ----------------
__global__ __launch_bounds__(256) void build_lists_kernel(const int* __restrict__ topk_id,
                                                          const float* __restrict__ topk_w,
                                                          int* __restrict__ list_idx,
                                                          float* __restrict__ list_w,
                                                          int* __restrict__ counts) {
  const int e = blockIdx.x;
  const int tid = threadIdx.x;
  __shared__ int scan[256];
  __shared__ int running;
  if (tid == 0) running = 0;
  __syncthreads();

  for (int t0 = 0; t0 < T_TOK; t0 += 256) {
    int t = t0 + tid;
    int id0 = topk_id[2 * t], id1 = topk_id[2 * t + 1];
    int match = 0, k = 0;
    if (id0 == e) { match = 1; k = 0; }
    else if (id1 == e) { match = 1; k = 1; }
    scan[tid] = match;
    __syncthreads();
    for (int off = 1; off < 256; off <<= 1) {
      int v = (tid >= off) ? scan[tid - off] : 0;
      __syncthreads();
      scan[tid] += v;
      __syncthreads();
    }
    int excl = scan[tid] - match;
    int total = scan[255];
    if (match) {
      int pos = running + excl;
      list_idx[e * T_TOK + pos] = t * 2 + k;
      list_w[e * T_TOK + pos]   = topk_w[2 * t + k];
    }
    __syncthreads();
    if (tid == 0) running += total;
    __syncthreads();
  }
  if (tid == 0) counts[e] = running;
}

// ---------------- shared expert: M=32 tile, 512 threads (16 waves x 64 cols) ------
__global__ __launch_bounds__(512) void shared_expert_kernel(const float* __restrict__ x,
                                                            const __bf16* __restrict__ wg,
                                                            const __bf16* __restrict__ wu,
                                                            const __bf16* __restrict__ wd,
                                                            float* __restrict__ out) {
  __shared__ __align__(16) __bf16 smem[32 * D_H];   // 64KB: x-tile, reused as sh-tile
  const int tid  = threadIdx.x;
  const int lane = tid & 31;
  const int wid  = tid >> 5;          // 0..15
  const int half = lane >> 4;
  const int l15  = lane & 15;
  const int t0   = blockIdx.x * 32;

  // load 32x1024 x-tile, fp32 -> bf16, swizzled
  for (int i = tid; i < 32 * (D_H / 4); i += 512) {
    int r = i >> 8, c4 = i & 255;
    float4 v = ((const float4*)(x + (size_t)(t0 + r) * D_H))[c4];
    __bf16* p = &smem[sw_idx(r, c4 * 4, D_H)];
    p[0] = f32_to_bf16(v.x); p[1] = f32_to_bf16(v.y);
    p[2] = f32_to_bf16(v.z); p[3] = f32_to_bf16(v.w);
  }
  __syncthreads();

  const v8f vzero = {0.f, 0.f, 0.f, 0.f, 0.f, 0.f, 0.f, 0.f};
  const int cb = wid * 64;                       // this wave's N columns
  v8f cg0[4], cg1[4], cu0[4], cu1[4];
#pragma unroll
  for (int j = 0; j < 4; ++j) { cg0[j] = vzero; cg1[j] = vzero; cu0[j] = vzero; cu1[j] = vzero; }

  for (int k = 0; k < D_H; k += 32) {
    v16bf a0 = load_a_sw(smem, l15,      D_H, k, half);
    v16bf a1 = load_a_sw(smem, l15 + 16, D_H, k, half);
    v16bf bg[4], bu[4];
#pragma unroll
    for (int j = 0; j < 4; ++j) {
      size_t roff = (size_t)(cb + j * 16 + l15) * D_H + k + 16 * half;
      bg[j] = load_b_frag(wg + roff);
      bu[j] = load_b_frag(wu + roff);
      if (k + 64 < D_H) {                           // global_prefetch_b8
        __builtin_prefetch((const void*)(wg + roff + 64), 0, 3);
        __builtin_prefetch((const void*)(wu + roff + 64), 0, 3);
      }
    }
#pragma unroll
    for (int j = 0; j < 4; ++j) {
      cg0[j] = wmma_bf16(a0, bg[j], cg0[j]);
      cg1[j] = wmma_bf16(a1, bg[j], cg1[j]);
      cu0[j] = wmma_bf16(a0, bu[j], cu0[j]);
      cu1[j] = wmma_bf16(a1, bu[j], cu1[j]);
    }
  }
  __syncthreads();   // everyone done with x-tile before overwriting with sh

#pragma unroll
  for (int j = 0; j < 4; ++j) {
#pragma unroll
    for (int r = 0; r < 8; ++r) {
      int m = r + 8 * half;
      int col = cb + j * 16 + l15;
      smem[sw_idx(m,      col, D_H)] = f32_to_bf16(silu(cg0[j][r]) * cu0[j][r]);
      smem[sw_idx(m + 16, col, D_H)] = f32_to_bf16(silu(cg1[j][r]) * cu1[j][r]);
    }
  }
  __syncthreads();

  // down: A = sh (32 x 1024), B = ws_down [1024 x 1024]
  v8f co0[4], co1[4];
#pragma unroll
  for (int j = 0; j < 4; ++j) { co0[j] = vzero; co1[j] = vzero; }
  for (int k = 0; k < D_SE; k += 32) {
    v16bf a0 = load_a_sw(smem, l15,      D_H, k, half);
    v16bf a1 = load_a_sw(smem, l15 + 16, D_H, k, half);
    v16bf bd[4];
#pragma unroll
    for (int j = 0; j < 4; ++j) {
      size_t roff = (size_t)(cb + j * 16 + l15) * D_SE + k + 16 * half;
      bd[j] = load_b_frag(wd + roff);
      if (k + 64 < D_SE) __builtin_prefetch((const void*)(wd + roff + 64), 0, 3);
    }
#pragma unroll
    for (int j = 0; j < 4; ++j) {
      co0[j] = wmma_bf16(a0, bd[j], co0[j]);
      co1[j] = wmma_bf16(a1, bd[j], co1[j]);
    }
  }
#pragma unroll
  for (int j = 0; j < 4; ++j) {
#pragma unroll
    for (int r = 0; r < 8; ++r) {
      int m = r + 8 * half;
      int col = cb + j * 16 + l15;
      out[(size_t)(t0 + m) * D_H + col]      = co0[j][r];
      out[(size_t)(t0 + m + 16) * D_H + col] = co1[j][r];
    }
  }
}

// ---------------- routed experts: M=32 tile, 256 threads (8 waves) ----------------
__global__ __launch_bounds__(256) void routed_expert_kernel(const float* __restrict__ x,
                                                            const __bf16* __restrict__ wg,
                                                            const __bf16* __restrict__ wu,
                                                            const __bf16* __restrict__ wd,
                                                            const int* __restrict__ counts,
                                                            const int* __restrict__ list_idx,
                                                            const float* __restrict__ list_w,
                                                            float* __restrict__ rb) {
  const int e    = blockIdx.y;
  const int tile = blockIdx.x;
  const int cnt  = counts[e];
  if (tile * 32 >= cnt) return;

  __shared__ __align__(16) __bf16 smem[32 * D_H];   // 64KB: x-tile, reused as act-tile
  const int tid  = threadIdx.x;
  const int lane = tid & 31;
  const int wid  = tid >> 5;          // 0..7
  const int half = lane >> 4;
  const int l15  = lane & 15;
  const int gbase = tile * 32;
  const int* lidx = list_idx + e * T_TOK;
  const float* lw = list_w + e * T_TOK;

  // gather 32 token rows, fp32 -> bf16, swizzled
  for (int i = tid; i < 32 * (D_H / 4); i += 256) {
    int r = i >> 8, c4 = i & 255;
    int g = gbase + r;
    int t = (g < cnt) ? (lidx[g] >> 1) : 0;
    float4 v = ((const float4*)(x + (size_t)t * D_H))[c4];
    __bf16* p = &smem[sw_idx(r, c4 * 4, D_H)];
    p[0] = f32_to_bf16(v.x); p[1] = f32_to_bf16(v.y);
    p[2] = f32_to_bf16(v.z); p[3] = f32_to_bf16(v.w);
  }
  __syncthreads();

  const __bf16* wge = wg + (size_t)e * D_E * D_H;
  const __bf16* wue = wu + (size_t)e * D_E * D_H;
  const __bf16* wde = wd + (size_t)e * D_H * D_E;
  const v8f vzero = {0.f, 0.f, 0.f, 0.f, 0.f, 0.f, 0.f, 0.f};

  // stage 1: gate/up, N = 512 -> wave owns 64 cols (4 tiles), M = 32
  v8f cg0[4], cg1[4], cu0[4], cu1[4];
#pragma unroll
  for (int j = 0; j < 4; ++j) { cg0[j] = vzero; cg1[j] = vzero; cu0[j] = vzero; cu1[j] = vzero; }
  const int cb1 = wid * 64;
  for (int k = 0; k < D_H; k += 32) {
    v16bf a0 = load_a_sw(smem, l15,      D_H, k, half);
    v16bf a1 = load_a_sw(smem, l15 + 16, D_H, k, half);
    v16bf bg[4], bu[4];
#pragma unroll
    for (int j = 0; j < 4; ++j) {
      size_t roff = (size_t)(cb1 + j * 16 + l15) * D_H + k + 16 * half;
      bg[j] = load_b_frag(wge + roff);
      bu[j] = load_b_frag(wue + roff);
      if (k + 64 < D_H) {
        __builtin_prefetch((const void*)(wge + roff + 64), 0, 3);
        __builtin_prefetch((const void*)(wue + roff + 64), 0, 3);
      }
    }
#pragma unroll
    for (int j = 0; j < 4; ++j) {
      cg0[j] = wmma_bf16(a0, bg[j], cg0[j]);
      cg1[j] = wmma_bf16(a1, bg[j], cg1[j]);
      cu0[j] = wmma_bf16(a0, bu[j], cu0[j]);
      cu1[j] = wmma_bf16(a1, bu[j], cu1[j]);
    }
  }
  __syncthreads();   // done reading x-tile

  // act = silu(g)*u*combine  -> bf16 tile [32 x 512] (reuses smem, stride 512)
  float rw0[8], rw1[8];
#pragma unroll
  for (int r = 0; r < 8; ++r) {
    int g0 = gbase + r + 8 * half;
    int g1 = g0 + 16;
    rw0[r] = (g0 < cnt) ? lw[g0] : 0.0f;
    rw1[r] = (g1 < cnt) ? lw[g1] : 0.0f;
  }
#pragma unroll
  for (int j = 0; j < 4; ++j) {
#pragma unroll
    for (int r = 0; r < 8; ++r) {
      int m = r + 8 * half;
      int col = cb1 + j * 16 + l15;
      smem[sw_idx(m,      col, D_E)] = f32_to_bf16(silu(cg0[j][r]) * cu0[j][r] * rw0[r]);
      smem[sw_idx(m + 16, col, D_E)] = f32_to_bf16(silu(cg1[j][r]) * cu1[j][r] * rw1[r]);
    }
  }
  __syncthreads();

  // stage 2: down, K = 512, N = 1024 -> wave owns 128 cols (8 tiles), M = 32
  v8f co0[8], co1[8];
#pragma unroll
  for (int j = 0; j < 8; ++j) { co0[j] = vzero; co1[j] = vzero; }
  const int cb2 = wid * 128;
  for (int k = 0; k < D_E; k += 32) {
    v16bf a0 = load_a_sw(smem, l15,      D_E, k, half);
    v16bf a1 = load_a_sw(smem, l15 + 16, D_E, k, half);
    v16bf bd[8];
#pragma unroll
    for (int j = 0; j < 8; ++j) {
      size_t roff = (size_t)(cb2 + j * 16 + l15) * D_E + k + 16 * half;
      bd[j] = load_b_frag(wde + roff);
      if (k + 64 < D_E) __builtin_prefetch((const void*)(wde + roff + 64), 0, 3);
    }
#pragma unroll
    for (int j = 0; j < 8; ++j) {
      co0[j] = wmma_bf16(a0, bd[j], co0[j]);
      co1[j] = wmma_bf16(a1, bd[j], co1[j]);
    }
  }
  // store to per-(token,slot) buffer
  int id0[8], id1[8];
#pragma unroll
  for (int r = 0; r < 8; ++r) {
    int g0 = gbase + r + 8 * half;
    int g1 = g0 + 16;
    id0[r] = (g0 < cnt) ? lidx[g0] : -1;
    id1[r] = (g1 < cnt) ? lidx[g1] : -1;
  }
#pragma unroll
  for (int j = 0; j < 8; ++j) {
#pragma unroll
    for (int r = 0; r < 8; ++r) {
      int col = cb2 + j * 16 + l15;
      if (id0[r] >= 0) rb[(size_t)id0[r] * D_H + col] = co0[j][r];
      if (id1[r] >= 0) rb[(size_t)id1[r] * D_H + col] = co1[j][r];
    }
  }
}

// ---------------- combine: out += rb[t,0] + rb[t,1] (fixed order) ----------------
__global__ __launch_bounds__(256) void combine_kernel(const float* __restrict__ rb,
                                                      float* __restrict__ out) {
  int i = blockIdx.x * 256 + threadIdx.x;
  if (i >= T_TOK * (D_H / 4)) return;
  int t = i >> 8, c4 = i & 255;
  float4 a  = ((const float4*)out)[i];
  float4 r0 = ((const float4*)(rb + (size_t)(t * 2 + 0) * D_H))[c4];
  float4 r1 = ((const float4*)(rb + (size_t)(t * 2 + 1) * D_H))[c4];
  a.x += r0.x + r1.x; a.y += r0.y + r1.y;
  a.z += r0.z + r1.z; a.w += r0.w + r1.w;
  ((float4*)out)[i] = a;
}

// ---------------- host ----------------
extern "C" void kernel_launch(void* const* d_in, const int* in_sizes, int n_in,
                              void* d_out, int out_size, void* d_ws, size_t ws_size,
                              hipStream_t stream) {
  (void)in_sizes; (void)n_in; (void)out_size; (void)ws_size;
  const float* x         = (const float*)d_in[0];
  const float* wg_router = (const float*)d_in[1];
  const float* w_gate    = (const float*)d_in[2];
  const float* w_up      = (const float*)d_in[3];
  const float* w_down    = (const float*)d_in[4];
  const float* ws_gate   = (const float*)d_in[5];
  const float* ws_up     = (const float*)d_in[6];
  const float* ws_down   = (const float*)d_in[7];
  float* out = (float*)d_out;

  char* ws = (char*)d_ws;
  size_t off = 0;
  auto take = [&](size_t bytes) -> char* {
    char* p = ws + off;
    off = (off + bytes + 255) & ~(size_t)255;
    return p;
  };
  const size_t NW_R = (size_t)N_EXP * D_E * D_H;
  const size_t NW_S = (size_t)D_SE * D_H;
  __bf16* wb_gate  = (__bf16*)take(NW_R * 2);
  __bf16* wb_up    = (__bf16*)take(NW_R * 2);
  __bf16* wb_down  = (__bf16*)take(NW_R * 2);
  __bf16* wsb_gate = (__bf16*)take(NW_S * 2);
  __bf16* wsb_up   = (__bf16*)take(NW_S * 2);
  __bf16* wsb_down = (__bf16*)take(NW_S * 2);
  int*   topk_id   = (int*)  take((size_t)T_TOK * 2 * 4);
  float* topk_w    = (float*)take((size_t)T_TOK * 2 * 4);
  int*   counts    = (int*)  take(N_EXP * 4);
  int*   list_idx  = (int*)  take((size_t)N_EXP * T_TOK * 4);
  float* list_w    = (float*)take((size_t)N_EXP * T_TOK * 4);
  float* rb        = (float*)take((size_t)T_TOK * 2 * D_H * 4);

  {
    int n4r = (int)(NW_R / 4), n4s = (int)(NW_S / 4);
    cvt_bf16_kernel<<<(n4r + 255) / 256, 256, 0, stream>>>(w_gate,  wb_gate,  n4r);
    cvt_bf16_kernel<<<(n4r + 255) / 256, 256, 0, stream>>>(w_up,    wb_up,    n4r);
    cvt_bf16_kernel<<<(n4r + 255) / 256, 256, 0, stream>>>(w_down,  wb_down,  n4r);
    cvt_bf16_kernel<<<(n4s + 255) / 256, 256, 0, stream>>>(ws_gate, wsb_gate, n4s);
    cvt_bf16_kernel<<<(n4s + 255) / 256, 256, 0, stream>>>(ws_up,   wsb_up,   n4s);
    cvt_bf16_kernel<<<(n4s + 255) / 256, 256, 0, stream>>>(ws_down, wsb_down, n4s);
  }
  router_kernel<<<T_TOK / 8, 256, 0, stream>>>(x, wg_router, topk_id, topk_w);
  build_lists_kernel<<<N_EXP, 256, 0, stream>>>(topk_id, topk_w, list_idx, list_w, counts);
  shared_expert_kernel<<<T_TOK / 32, 512, 0, stream>>>(x, wsb_gate, wsb_up, wsb_down, out);
  routed_expert_kernel<<<dim3(T_TOK / 32, N_EXP), 256, 0, stream>>>(
      x, wb_gate, wb_up, wb_down, counts, list_idx, list_w, rb);
  combine_kernel<<<(T_TOK * (D_H / 4) + 255) / 256, 256, 0, stream>>>(rb, out);
}
